// Short_Mem_74861279969416
// MI455X (gfx1250) — matrix operand order repaired and checked by reading, not compiled, not run on hardware
//
#include <hip/hip_runtime.h>
#include <hip/hip_bf16.h>
#include <math.h>
#include <stdint.h>

#define B_  4
#define T_  4096
#define C_  2048
#define M_  (B_*T_)      // 16384 rows
#define K1_ (2*C_)       // 4096
#define N1_ (2*C_)       // 4096 (act | rec)

typedef float  v8f   __attribute__((ext_vector_type(8)));
typedef __bf16 v16bf __attribute__((ext_vector_type(16)));
typedef int    v4i_  __attribute__((vector_size(16)));

typedef __attribute__((address_space(1))) v4i_* gptr_v4i;
typedef __attribute__((address_space(3))) v4i_* lptr_v4i;

union FragU { uint4 q[2]; v16bf v; };

#if defined(__has_builtin)
# if __has_builtin(__builtin_amdgcn_global_load_async_to_lds_b128)
#  define HAVE_ASYNC_LDS 1
# endif
#endif

// ---- conversions ----
__device__ __forceinline__ unsigned int f2bf2(float lo, float hi) {
  unsigned int ul = __float_as_uint(lo);
  unsigned int uh = __float_as_uint(hi);
  ul = (ul + 0x7FFFu + ((ul >> 16) & 1u)) >> 16;
  uh = (uh + 0x7FFFu + ((uh >> 16) & 1u)) >> 16;
  return (uh << 16) | (ul & 0xFFFFu);
}
__device__ __forceinline__ unsigned short f2bf(float f) {
  unsigned int u = __float_as_uint(f);
  return (unsigned short)((u + 0x7FFFu + ((u >> 16) & 1u)) >> 16);
}
__device__ __forceinline__ float bf2f(unsigned int h) {
  return __uint_as_float((h & 0xFFFFu) << 16);
}
__device__ __forceinline__ void cvt16(float4 a, float4 b, float4 c, float4 d,
                                      uint4* o0, uint4* o1) {
  o0->x = f2bf2(a.x, a.y); o0->y = f2bf2(a.z, a.w);
  o0->z = f2bf2(b.x, b.y); o0->w = f2bf2(b.z, b.w);
  o1->x = f2bf2(c.x, c.y); o1->y = f2bf2(c.z, c.w);
  o1->z = f2bf2(d.x, d.y); o1->w = f2bf2(d.z, d.w);
}
__device__ __forceinline__ float sigm(float v) {
  float z = v > 0.f ? v : 0.f;
  return 1.f / (1.f + __expf(-z));
}

// ---- async 16B copy global->LDS (gfx1250), with sync fallback ----
__device__ __forceinline__ void async_cp16(const unsigned short* gsrc, unsigned short* ldst) {
#ifdef HAVE_ASYNC_LDS
  gptr_v4i g = (gptr_v4i)(unsigned long long)(uintptr_t)gsrc;
  lptr_v4i l = (lptr_v4i)(unsigned int)(uintptr_t)ldst;
  __builtin_amdgcn_global_load_async_to_lds_b128(g, l, 0, 0);
#else
  *(uint4*)ldst = *(const uint4*)gsrc;
#endif
}
__device__ __forceinline__ void wait_async_le4() {
#ifdef HAVE_ASYNC_LDS
  asm volatile("s_wait_asynccnt 0x4" ::: "memory");
#endif
}
__device__ __forceinline__ void wait_async_0() {
#ifdef HAVE_ASYNC_LDS
  asm volatile("s_wait_asynccnt 0x0" ::: "memory");
#endif
}

// =========== conversion kernels (one streaming pass each) ===========
// xb[B][T+1][C]: row tt==0 is zeros, row tt>=1 is x[tt-1]  -> TimeShift by indexing
__global__ __launch_bounds__(256)
void conv_x(const float* __restrict__ x, unsigned short* __restrict__ xb) {
  const int row = blockIdx.x;            // 0 .. B*(T+1)-1
  const int tt  = row % (T_ + 1);
  const int b   = row / (T_ + 1);
  const int ci  = threadIdx.x * 8;
  uint4 o = make_uint4(0u, 0u, 0u, 0u);
  if (tt > 0) {
    const float4* p = (const float4*)(x + (size_t)(b * T_ + tt - 1) * C_ + ci);
    float4 a = p[0], bq = p[1];
    o.x = f2bf2(a.x, a.y); o.y = f2bf2(a.z, a.w);
    o.z = f2bf2(bq.x, bq.y); o.w = f2bf2(bq.z, bq.w);
  }
  *(uint4*)(xb + (size_t)row * C_ + ci) = o;
}

// wcat[4096][4096] = [W_act ; W_rec] rows, bf16
__global__ __launch_bounds__(256)
void conv_w(const float* __restrict__ Wact, const float* __restrict__ Wrec,
            unsigned short* __restrict__ wcat) {
  const int n  = blockIdx.x;
  const int ci = threadIdx.x * 16;
  const float* src = (n < C_) ? (Wact + (size_t)n * K1_) : (Wrec + (size_t)(n - C_) * K1_);
  const float4* p = (const float4*)(src + ci);
  float4 a = p[0], b = p[1], c = p[2], d = p[3];
  uint4 q0, q1; cvt16(a, b, c, d, &q0, &q1);
  *(uint4*)(wcat + (size_t)n * K1_ + ci)     = q0;
  *(uint4*)(wcat + (size_t)n * K1_ + ci + 8) = q1;
}

__global__ __launch_bounds__(256)
void conv_wv(const float* __restrict__ Wval, unsigned short* __restrict__ wvb) {
  const int n  = blockIdx.x;
  const int ci = threadIdx.x * 8;
  const float4* p = (const float4*)(Wval + (size_t)n * C_ + ci);
  float4 a = p[0], b = p[1];
  uint4 o;
  o.x = f2bf2(a.x, a.y); o.y = f2bf2(a.z, a.w);
  o.z = f2bf2(b.x, b.y); o.w = f2bf2(b.z, b.w);
  *(uint4*)(wvb + (size_t)n * C_ + ci) = o;
}

// =========== Kernel 1: ct @ [W_act|W_rec]^T  (async double-buffered WMMA) ===========
__global__ __launch_bounds__(256)
void gemm_act_rec(const unsigned short* __restrict__ xb,     // [B][T+1][C] bf16
                  const unsigned short* __restrict__ wcat,   // [4096][4096] bf16
                  unsigned short* __restrict__ xvb,          // [M_,C_] bf16
                  unsigned short* __restrict__ r4b) {        // [M_,C_] bf16
  __shared__ __align__(16) unsigned short lsA[2][128 * 32];
  __shared__ __align__(16) unsigned short lsB[2][128 * 32];
  const int tid   = threadIdx.x;
  const int lane  = tid & 31;
  const int wv    = tid >> 5;
  const int waveM = wv & 1;
  const int waveN = wv >> 1;
  const int rowL  = tid >> 1;
  const int half  = tid & 1;

  const int gRow0 = blockIdx.y * 128;
  const int gCol0 = blockIdx.x * 128;
  const int grow  = gRow0 + rowL;
  const int bb    = grow >> 12;
  const int t     = grow & (T_ - 1);

  const unsigned short* aShift = xb + (size_t)(bb * (T_ + 1) + t) * C_;       // k <  C
  const unsigned short* aCur   = xb + (size_t)(bb * (T_ + 1) + t + 1) * C_;   // k >= C
  const unsigned short* wrow   = wcat + (size_t)(gCol0 + rowL) * K1_;

  v8f acc[4][2] = {};

  auto issue = [&](int ks, int buf) {
    const int kb = ks * 32 + half * 16;
    const unsigned short* gA = (kb < C_) ? (aShift + kb) : (aCur + (kb - C_));
    unsigned short* dA = &lsA[buf][rowL * 32 + half * 16];
    async_cp16(gA,     dA);
    async_cp16(gA + 8, dA + 8);
    const unsigned short* gB = wrow + kb;
    unsigned short* dB = &lsB[buf][rowL * 32 + half * 16];
    async_cp16(gB,     dB);
    async_cp16(gB + 8, dB + 8);
  };

  const int NK = K1_ / 32;
  issue(0, 0);
  for (int ks = 0; ks < NK; ++ks) {
    const int buf = ks & 1;
    if (ks + 1 < NK) { issue(ks + 1, buf ^ 1); wait_async_le4(); }
    else             { wait_async_0(); }
    __syncthreads();

    const int kh  = lane >> 4;
    const int l15 = lane & 15;
    FragU af[4];
#pragma unroll
    for (int m = 0; m < 4; ++m) {
      int r = waveM * 64 + m * 16 + l15;
      af[m].q[0] = *(const uint4*)&lsA[buf][r * 32 + kh * 8];
      af[m].q[1] = *(const uint4*)&lsA[buf][r * 32 + kh * 8 + 16];
    }
    FragU bfg[2];
#pragma unroll
    for (int n = 0; n < 2; ++n) {
      int cc = waveN * 32 + n * 16 + l15;
      bfg[n].q[0] = *(const uint4*)&lsB[buf][cc * 32 + kh * 8];
      bfg[n].q[1] = *(const uint4*)&lsB[buf][cc * 32 + kh * 8 + 16];
    }
#pragma unroll
    for (int m = 0; m < 4; ++m)
#pragma unroll
      for (int n = 0; n < 2; ++n)
        acc[m][n] = __builtin_amdgcn_wmma_f32_16x16x32_bf16(
            false, af[m].v, false, bfg[n].v, (short)0, acc[m][n], false, false);
    __syncthreads();
  }

  const int l15 = lane & 15;
  const int hi8 = (lane >> 4) * 8;
#pragma unroll
  for (int m = 0; m < 4; ++m)
#pragma unroll
    for (int n = 0; n < 2; ++n)
#pragma unroll
      for (int i = 0; i < 8; ++i) {
        int row = gRow0 + waveM * 64 + m * 16 + hi8 + i;
        int col = gCol0 + waveN * 32 + n * 16 + l15;
        float s = sigm(acc[m][n][i]);
        if (col < C_) xvb[(size_t)row * C_ + col] = f2bf(s);
        else { float s2 = s * s; r4b[(size_t)row * C_ + (col - C_)] = f2bf(s2 * s2); }
      }
}

// =========== Kernel 2: cummax-normalize + complex cumprod along T ===========
__global__ __launch_bounds__(256)
void cscan(const unsigned short* __restrict__ xvb, unsigned short* __restrict__ kvb) {
  __shared__ float s[128 * 20];           // 128 t x 16 ch, padded
  const int tid  = threadIdx.x;
  const int lane = tid & 31;
  const int wv   = tid >> 5;              // complex pair 0..7
  const int g    = blockIdx.x & 127;
  const int bb   = blockIdx.x >> 7;
  const int c0   = g * 16;
  const int tl   = tid >> 1;
  const int half = tid & 1;

  float mcar = 0.f, cr = 1.f, ci = 0.f;

  for (int tb = 0; tb < T_; tb += 128) {
    {   // coalesced bf16 load + unpack to f32 LDS
      const uint4 u = *(const uint4*)(xvb + (size_t)(bb * T_ + tb + tl) * C_ + c0 + half * 8);
      float* d = &s[tl * 20 + half * 8];
      d[0] = bf2f(u.x); d[1] = bf2f(u.x >> 16);
      d[2] = bf2f(u.y); d[3] = bf2f(u.y >> 16);
      d[4] = bf2f(u.z); d[5] = bf2f(u.z >> 16);
      d[6] = bf2f(u.w); d[7] = bf2f(u.w >> 16);
    }
    __syncthreads();
#pragma unroll
    for (int sub = 0; sub < 4; ++sub) {
      int t_l = sub * 32 + lane;
      float re = s[t_l * 20 + 2 * wv];
      float im = s[t_l * 20 + 2 * wv + 1];
      float mm = sqrtf(re * re + im * im);
#pragma unroll
      for (int d = 1; d < 32; d <<= 1) {            // inclusive cummax
        float o = __shfl_up(mm, d, 32);
        if (lane >= d) mm = fmaxf(mm, o);
      }
      mm = fmaxf(mm, mcar);
      float pr = re / mm, pi = im / mm;
#pragma unroll
      for (int d = 1; d < 32; d <<= 1) {            // inclusive complex cumprod
        float orr = __shfl_up(pr, d, 32);
        float oii = __shfl_up(pi, d, 32);
        if (lane >= d) {
          float nr = pr * orr - pi * oii;
          float ni = pr * oii + pi * orr;
          pr = nr; pi = ni;
        }
      }
      float fr = cr * pr - ci * pi;
      float fi = cr * pi + ci * pr;
      s[t_l * 20 + 2 * wv]     = fr;
      s[t_l * 20 + 2 * wv + 1] = fi;
      mcar = __shfl(mm, 31, 32);
      float t0 = __shfl(fr, 31, 32);
      float t1 = __shfl(fi, 31, 32);
      cr = t0; ci = t1;
    }
    __syncthreads();
    {   // pack f32 -> bf16, coalesced store
      const float* d = &s[tl * 20 + half * 8];
      uint4 o;
      o.x = f2bf2(d[0], d[1]); o.y = f2bf2(d[2], d[3]);
      o.z = f2bf2(d[4], d[5]); o.w = f2bf2(d[6], d[7]);
      *(uint4*)(kvb + (size_t)(bb * T_ + tb + tl) * C_ + c0 + half * 8) = o;
    }
    __syncthreads();
  }
}

// =========== Kernel 3: kv @ W_val^T, gated by r^4 ===========
__global__ __launch_bounds__(256)
void gemm_out(const unsigned short* __restrict__ kvb,   // [M_,C_] bf16
              const unsigned short* __restrict__ wvb,   // [C_,C_] bf16
              const unsigned short* __restrict__ r4b,   // [M_,C_] bf16
              float* __restrict__ out) {
  __shared__ __align__(16) unsigned short lsA[2][128 * 32];
  __shared__ __align__(16) unsigned short lsB[2][128 * 32];
  const int tid   = threadIdx.x;
  const int lane  = tid & 31;
  const int wv    = tid >> 5;
  const int waveM = wv & 1;
  const int waveN = wv >> 1;
  const int rowL  = tid >> 1;
  const int half  = tid & 1;

  const int gRow0 = blockIdx.y * 128;
  const int gCol0 = blockIdx.x * 128;
  const unsigned short* abase = kvb + (size_t)(gRow0 + rowL) * C_;
  const unsigned short* wrow  = wvb + (size_t)(gCol0 + rowL) * C_;

  v8f acc[4][2] = {};

  auto issue = [&](int ks, int buf) {
    const int kb = ks * 32 + half * 16;
    unsigned short* dA = &lsA[buf][rowL * 32 + half * 16];
    async_cp16(abase + kb,     dA);
    async_cp16(abase + kb + 8, dA + 8);
    unsigned short* dB = &lsB[buf][rowL * 32 + half * 16];
    async_cp16(wrow + kb,     dB);
    async_cp16(wrow + kb + 8, dB + 8);
  };

  const int NK = C_ / 32;
  issue(0, 0);
  for (int ks = 0; ks < NK; ++ks) {
    const int buf = ks & 1;
    if (ks + 1 < NK) { issue(ks + 1, buf ^ 1); wait_async_le4(); }
    else             { wait_async_0(); }
    __syncthreads();

    const int kh  = lane >> 4;
    const int l15 = lane & 15;
    FragU af[4];
#pragma unroll
    for (int m = 0; m < 4; ++m) {
      int r = waveM * 64 + m * 16 + l15;
      af[m].q[0] = *(const uint4*)&lsA[buf][r * 32 + kh * 8];
      af[m].q[1] = *(const uint4*)&lsA[buf][r * 32 + kh * 8 + 16];
    }
    FragU bfg[2];
#pragma unroll
    for (int n = 0; n < 2; ++n) {
      int cc = waveN * 32 + n * 16 + l15;
      bfg[n].q[0] = *(const uint4*)&lsB[buf][cc * 32 + kh * 8];
      bfg[n].q[1] = *(const uint4*)&lsB[buf][cc * 32 + kh * 8 + 16];
    }
#pragma unroll
    for (int m = 0; m < 4; ++m)
#pragma unroll
      for (int n = 0; n < 2; ++n)
        acc[m][n] = __builtin_amdgcn_wmma_f32_16x16x32_bf16(
            false, af[m].v, false, bfg[n].v, (short)0, acc[m][n], false, false);
    __syncthreads();
  }

  const int l15 = lane & 15;
  const int hi8 = (lane >> 4) * 8;
#pragma unroll
  for (int m = 0; m < 4; ++m)
#pragma unroll
    for (int n = 0; n < 2; ++n)
#pragma unroll
      for (int i = 0; i < 8; ++i) {
        int row = gRow0 + waveM * 64 + m * 16 + hi8 + i;
        int col = gCol0 + waveN * 32 + n * 16 + l15;
        size_t idx = (size_t)row * C_ + col;
        out[idx] = acc[m][n][i] * bf2f((unsigned int)r4b[idx]);
      }
}

extern "C" void kernel_launch(void* const* d_in, const int* in_sizes, int n_in,
                              void* d_out, int out_size, void* d_ws, size_t ws_size,
                              hipStream_t stream) {
  const float* x    = (const float*)d_in[0];
  const float* Wact = (const float*)d_in[1];
  const float* Wval = (const float*)d_in[2];
  const float* Wrec = (const float*)d_in[3];

  unsigned short* ws = (unsigned short*)d_ws;
  const size_t nMC = (size_t)M_ * C_;
  unsigned short* xvb  = ws;                                  // [M_,C_]
  unsigned short* r4b  = ws + nMC;                            // [M_,C_]
  unsigned short* kvb  = ws + 2 * nMC;                        // [M_,C_]
  unsigned short* xb   = ws + 3 * nMC;                        // [B][T+1][C]
  const size_t nXB = (size_t)B_ * (T_ + 1) * C_;
  unsigned short* wcat = xb + nXB;                            // [4096][4096]
  unsigned short* wvb  = wcat + (size_t)N1_ * K1_;            // [2048][2048]

  conv_x <<<B_ * (T_ + 1), 256, 0, stream>>>(x, xb);
  conv_w <<<N1_,           256, 0, stream>>>(Wact, Wrec, wcat);
  conv_wv<<<C_,            256, 0, stream>>>(Wval, wvb);

  dim3 g1(N1_ / 128, M_ / 128);
  gemm_act_rec<<<g1, 256, 0, stream>>>(xb, wcat, xvb, r4b);

  cscan<<<(B_ * (C_ / 2)) / 8, 256, 0, stream>>>(xvb, kvb);

  dim3 g3(C_ / 128, M_ / 128);
  gemm_out<<<g3, 256, 0, stream>>>(kvb, wvb, r4b, (float*)d_out);
}
